// CBANet_29240137351591
// MI455X (gfx1250) — matrix operand order repaired and checked by reading, not compiled, not run on hardware
//
#include <hip/hip_runtime.h>
#include <hip/hip_bf16.h>
#include <math.h>

typedef __attribute__((ext_vector_type(16))) _Float16 v16h;
typedef __attribute__((ext_vector_type(8)))  float    v8f;

#define GM_STORE   0
#define GM_RELU    1
#define GM_ROWMAX  2
#define GM_SIGMOID 3

#define NB 2
#define NPTS 12000
#define MSEL 6000
#define QS 64
#define KNN 16
#define WPB 4   // waves per block in GEMM

// ---------------------------------------------------------------------------
// WMMA GEMM: Out[b] = act(W[b?]*X[b] + bias). One wave = one 16x16 tile.
// TRANSA: A[m][k] = W[k*ldA + m] (for mf = g_feats^T * mask_feats)
// KGUARD: C not a multiple of 32 (bb1 only) -> zero-fill k >= C.
// Rows/cols out of range are handled by ADDRESS CLAMPING + epilogue masking:
// garbage along M/N only reaches outputs that are masked at store (or zeroed
// before the ROWMAX cross-lane max); only K-garbage would corrupt valid
// results, and the fast path has exact K. This keeps the inner loop branch-free.
// ---------------------------------------------------------------------------
template<int TRANSA, int KGUARD>
__global__ void gemm_wmma_t(const float* __restrict__ Wp, const float* __restrict__ Xp,
                            const float* __restrict__ bias, float* __restrict__ Out,
                            int O, int C, int Ncols, int ldX, int ldOut,
                            long sW, long sX, long sOut, int mode, int ldA)
{
  const int lane = threadIdx.x & 31;
  const int wv   = threadIdx.x >> 5;
  const int colT = (blockIdx.x * WPB + wv) << 4;
  const int rowT = blockIdx.y << 4;
  const int b    = blockIdx.z;
  const float* Wb = Wp + (long)b * sW;
  const float* Xb = Xp + (long)b * sX;
  float* Ob = Out + (long)b * sOut;
  const int mloc  = lane & 15;
  const int half  = lane >> 4;
  const int arow  = rowT + mloc;
  const int arowc = min(arow, O - 1);      // clamped (safe) A row
  const int col   = colT + mloc;
  const int colc  = min(col, Ncols - 1);   // clamped (safe) X column
  v8f acc = {};
  for (int k0 = 0; k0 < C; k0 += 32) {
    v16h a, bv;
    if (KGUARD) {
#pragma unroll
      for (int j = 0; j < 8; ++j) {
        int ka = k0 + half * 8 + j, kb = ka + 16;
        float va = (ka < C) ? (TRANSA ? Wb[ka * ldA + arowc] : Wb[arowc * ldA + ka]) : 0.f;
        float vb = (kb < C) ? (TRANSA ? Wb[kb * ldA + arowc] : Wb[arowc * ldA + kb]) : 0.f;
        a[j] = (_Float16)va; a[8 + j] = (_Float16)vb;
      }
#pragma unroll
      for (int j = 0; j < 16; ++j) {
        int kx = k0 + half * 8 + j + (j >> 3) * 8;
        float v = (kx < C) ? Xb[kx * ldX + colc] : 0.f;
        bv[j] = (_Float16)v;
      }
    } else {
      if (TRANSA) {
#pragma unroll
        for (int j = 0; j < 8; ++j) {
          int ka = k0 + half * 8 + j;
          a[j]     = (_Float16)Wb[ka * ldA + arowc];
          a[8 + j] = (_Float16)Wb[(ka + 16) * ldA + arowc];
        }
      } else {
        const float4* Ar = reinterpret_cast<const float4*>(Wb + arowc * ldA + k0 + half * 8);
        float4 a0 = Ar[0], a1 = Ar[1], a2 = Ar[4], a3 = Ar[5];
        a[0]=(_Float16)a0.x; a[1]=(_Float16)a0.y; a[2] =(_Float16)a0.z; a[3] =(_Float16)a0.w;
        a[4]=(_Float16)a1.x; a[5]=(_Float16)a1.y; a[6] =(_Float16)a1.z; a[7] =(_Float16)a1.w;
        a[8]=(_Float16)a2.x; a[9]=(_Float16)a2.y; a[10]=(_Float16)a2.z; a[11]=(_Float16)a2.w;
        a[12]=(_Float16)a3.x; a[13]=(_Float16)a3.y; a[14]=(_Float16)a3.z; a[15]=(_Float16)a3.w;
      }
      const float* Xk = Xb + (k0 + half * 8) * ldX + colc;
#pragma unroll
      for (int j = 0; j < 16; ++j) bv[j] = (_Float16)Xk[(j + (j >> 3) * 8) * ldX];
    }
    acc = __builtin_amdgcn_wmma_f32_16x16x32_f16(false, a, false, bv, (short)0, acc, false, false);
  }
#pragma unroll
  for (int r = 0; r < 8; ++r) {
    int row = rowT + half * 8 + r;
    bool rowok = (row < O);
    float v = acc[r];
    if (rowok && bias) v += bias[row];
    if (mode == GM_RELU || mode == GM_ROWMAX) v = fmaxf(v, 0.f);
    else if (mode == GM_SIGMOID) v = 1.f / (1.f + expf(-v));
    if (mode == GM_ROWMAX) {
      if (!rowok || col >= Ncols) v = 0.f;
      v = fmaxf(v, __shfl_xor(v, 1));
      v = fmaxf(v, __shfl_xor(v, 2));
      v = fmaxf(v, __shfl_xor(v, 4));
      v = fmaxf(v, __shfl_xor(v, 8));
      if (mloc == 0 && rowok) atomicMax((unsigned int*)&Ob[row], __float_as_uint(v));
    } else if (rowok && col < Ncols) {
      Ob[row * ldOut + col] = v;
    }
  }
}

// ---------------------------------------------------------------------------
// EdgeConv fused: out[b][o][m] = max_k relu( sum_c W[o][c]*e[c][m*16+k] + b[o] )
// e on the fly: c<Cin -> F[c][idx[m][k]] - F[c][m]; c>=Cin -> F[c-Cin][m].
// One wave = 16(ch) x 16(neighbor) tile == one point; fused neighbor-max.
// O=256, C=512 exact -> no guards anywhere.
// ---------------------------------------------------------------------------
__global__ void edgeconv_wmma_kernel(const float* __restrict__ Wp, const float* __restrict__ bias,
                                     const float* __restrict__ F, const int* __restrict__ idx,
                                     float* __restrict__ Out,
                                     int Cin, int Mpts, int ldF,
                                     long sF, long sIdx, long sOut)
{
  const int lane = threadIdx.x & 31;
  const int m    = blockIdx.x;
  const int rowT = blockIdx.y << 4;
  const int b    = blockIdx.z;
  const float* Fb = F + (long)b * sF;
  const int* ib   = idx + (long)b * sIdx + m * KNN;
  const int mloc = lane & 15;
  const int half = lane >> 4;
  const int arow = rowT + mloc;
  const int C = 2 * Cin;
  const int nb = ib[mloc];
  v8f acc = {};
  for (int k0 = 0; k0 < C; k0 += 32) {
    v16h a, bv;
    const float4* Ar = reinterpret_cast<const float4*>(Wp + arow * C + k0 + half * 8);
    float4 a0 = Ar[0], a1 = Ar[1], a2 = Ar[4], a3 = Ar[5];
    a[0]=(_Float16)a0.x; a[1]=(_Float16)a0.y; a[2] =(_Float16)a0.z; a[3] =(_Float16)a0.w;
    a[4]=(_Float16)a1.x; a[5]=(_Float16)a1.y; a[6] =(_Float16)a1.z; a[7] =(_Float16)a1.w;
    a[8]=(_Float16)a2.x; a[9]=(_Float16)a2.y; a[10]=(_Float16)a2.z; a[11]=(_Float16)a2.w;
    a[12]=(_Float16)a3.x; a[13]=(_Float16)a3.y; a[14]=(_Float16)a3.z; a[15]=(_Float16)a3.w;
#pragma unroll
    for (int j = 0; j < 16; ++j) {
      int c = k0 + half * 8 + j + (j >> 3) * 8;
      float v;
      if (c < Cin) v = Fb[c * ldF + nb] - Fb[c * ldF + m];
      else         v = Fb[(c - Cin) * ldF + m];
      bv[j] = (_Float16)v;
    }
    acc = __builtin_amdgcn_wmma_f32_16x16x32_f16(false, a, false, bv, (short)0, acc, false, false);
  }
#pragma unroll
  for (int r = 0; r < 8; ++r) {
    int row = rowT + half * 8 + r;
    float v = fmaxf(acc[r] + bias[row], 0.f);
    v = fmaxf(v, __shfl_xor(v, 1));
    v = fmaxf(v, __shfl_xor(v, 2));
    v = fmaxf(v, __shfl_xor(v, 4));
    v = fmaxf(v, __shfl_xor(v, 8));
    if (mloc == 0) Out[(long)b * sOut + row * Mpts + m] = v;
  }
}

// ---- helpers --------------------------------------------------------------
__global__ void zero_f32_kernel(float* p, long n) {
  long i = blockIdx.x * (long)blockDim.x + threadIdx.x;
  if (i < n) p[i] = 0.f;
}

__global__ void broadcast_g_kernel(const float* __restrict__ g, float* __restrict__ feats) {
  int b = blockIdx.z;
  long t = blockIdx.x * (long)blockDim.x + threadIdx.x;
  if (t >= (long)1024 * NPTS) return;
  int o = (int)(t / NPTS), n = (int)(t % NPTS);
  feats[(long)b * 1344 * NPTS + (320 + o) * NPTS + n] = g[b * 1024 + o];
}

// Exact top-M selection by rank counting (sorted, stable ties == lax.top_k).
__global__ void topk_rank_kernel(const float* __restrict__ dm, long sDm,
                                 int* __restrict__ tidx, float* __restrict__ dsel)
{
  int b = blockIdx.z;
  const float* d = dm + (long)b * sDm;
  int i = blockIdx.x * blockDim.x + threadIdx.x;
  float v = (i < NPTS) ? d[i] : 0.f;
  int rank = 0;
  __shared__ float tile[256];
  for (int t0 = 0; t0 < NPTS; t0 += 256) {
    int j = t0 + threadIdx.x;
    tile[threadIdx.x] = (j < NPTS) ? d[j] : -3.4e38f;
    __syncthreads();
    int lim = min(256, NPTS - t0);
    for (int jj = 0; jj < lim; ++jj) {
      float u = tile[jj];
      rank += (u > v) || (u == v && (t0 + jj) < i);
    }
    __syncthreads();
  }
  if (i < NPTS && rank < MSEL) { tidx[b * MSEL + rank] = i; dsel[b * MSEL + rank] = v; }
}

__global__ void gather_cols_kernel(const float* __restrict__ src, long sSrc, int ldSrc,
                                   const int* __restrict__ idx, long sIdx,
                                   float* __restrict__ dst, long sDst, int ldDst,
                                   int rows, int cols)
{
  int b = blockIdx.z;
  long t = blockIdx.x * (long)blockDim.x + threadIdx.x;
  if (t >= (long)rows * cols) return;
  int r = (int)(t / cols), c = (int)(t % cols);
  dst[(long)b * sDst + r * ldDst + c] =
      src[(long)b * sSrc + r * ldSrc + idx[(long)b * sIdx + c]];
}

__global__ void gather_allidx_kernel(const int* __restrict__ tidx, const int* __restrict__ sel,
                                     float* __restrict__ out)
{
  int b = blockIdx.z, q = threadIdx.x;
  out[b * QS + q] = (float)tidx[b * MSEL + sel[b * QS + q]];
}

// KNN: per-query insertion sort of K smallest distances over LDS-staged tiles.
__global__ void knn_kernel(const float* __restrict__ P, long sP, int ldP,
                           int Mq, int* __restrict__ outIdx, long sOut)
{
  int b = blockIdx.z;
  const float* Pb = P + (long)b * sP;
  int i = blockIdx.x * blockDim.x + threadIdx.x;
  float xi = 0.f, yi = 0.f, zi = 0.f;
  if (i < Mq) { xi = Pb[i]; yi = Pb[ldP + i]; zi = Pb[2 * ldP + i]; }
  float bd[KNN]; int bix[KNN];
#pragma unroll
  for (int s = 0; s < KNN; ++s) { bd[s] = 3.4e38f; bix[s] = 0; }
  __shared__ float sx[256], sy[256], sz[256];
  for (int t0 = 0; t0 < Mq; t0 += 256) {
    int j = t0 + threadIdx.x;
    sx[threadIdx.x] = (j < Mq) ? Pb[j] : 0.f;
    sy[threadIdx.x] = (j < Mq) ? Pb[ldP + j] : 0.f;
    sz[threadIdx.x] = (j < Mq) ? Pb[2 * ldP + j] : 0.f;
    __syncthreads();
    int lim = min(256, Mq - t0);
    for (int jj = 0; jj < lim; ++jj) {
      float dx = sx[jj] - xi, dy = sy[jj] - yi, dz = sz[jj] - zi;
      float d = dx * dx + dy * dy + dz * dz;
      if (d < bd[KNN - 1]) {
        bd[KNN - 1] = d; bix[KNN - 1] = t0 + jj;
#pragma unroll
        for (int s = KNN - 1; s > 0; --s) {
          if (bd[s] < bd[s - 1]) {
            float td = bd[s]; bd[s] = bd[s - 1]; bd[s - 1] = td;
            int ti = bix[s]; bix[s] = bix[s - 1]; bix[s - 1] = ti;
          }
        }
      }
    }
    __syncthreads();
  }
  if (i < Mq) {
#pragma unroll
    for (int s = 0; s < KNN; ++s) outIdx[(long)b * sOut + i * KNN + s] = bix[s];
  }
}

// Farthest point sampling: one 1024-thread block per batch, d[] in LDS.
__global__ void __launch_bounds__(1024) fps_kernel(const float* __restrict__ P, long sP, int ldP,
                                                   const float* __restrict__ dmsel, long sDm,
                                                   int* __restrict__ sel, long sSel)
{
  int b = blockIdx.x;
  const float* Pb = P + (long)b * sP;
  const float* dmb = dmsel + (long)b * sDm;
  int* selb = sel + (long)b * sSel;
  __shared__ float dd[MSEL];
  __shared__ float rv[1024];
  __shared__ int   ri[1024];
  int tid = threadIdx.x;
  float bv = -3.4e38f; int bidx = 0;
  for (int i = tid; i < MSEL; i += 1024) { float v = dmb[i]; if (v > bv) { bv = v; bidx = i; } }
  rv[tid] = bv; ri[tid] = bidx; __syncthreads();
  for (int s = 512; s > 0; s >>= 1) {
    if (tid < s) {
      if (rv[tid + s] > rv[tid] || (rv[tid + s] == rv[tid] && ri[tid + s] < ri[tid])) {
        rv[tid] = rv[tid + s]; ri[tid] = ri[tid + s];
      }
    }
    __syncthreads();
  }
  int start = ri[0];
  if (tid == 0) selb[0] = start;
  float sx_ = Pb[start], sy_ = Pb[ldP + start], sz_ = Pb[2 * ldP + start];
  for (int i = tid; i < MSEL; i += 1024) {
    float dx = Pb[i] - sx_, dy = Pb[ldP + i] - sy_, dz = Pb[2 * ldP + i] - sz_;
    dd[i] = dx * dx + dy * dy + dz * dz;
  }
  __syncthreads();
  for (int q = 1; q < QS; ++q) {
    float mv = -3.4e38f; int mi = 0;
    for (int i = tid; i < MSEL; i += 1024) { float v = dd[i]; if (v > mv) { mv = v; mi = i; } }
    rv[tid] = mv; ri[tid] = mi; __syncthreads();
    for (int s = 512; s > 0; s >>= 1) {
      if (tid < s) {
        if (rv[tid + s] > rv[tid] || (rv[tid + s] == rv[tid] && ri[tid + s] < ri[tid])) {
          rv[tid] = rv[tid + s]; ri[tid] = ri[tid + s];
        }
      }
      __syncthreads();
    }
    int ci = ri[0];
    if (tid == 0) selb[q] = ci;
    float cx_ = Pb[ci], cy_ = Pb[ldP + ci], cz_ = Pb[2 * ldP + ci];
    __syncthreads();
    for (int i = tid; i < MSEL; i += 1024) {
      float dx = Pb[i] - cx_, dy = Pb[ldP + i] - cy_, dz = Pb[2 * ldP + i] - cz_;
      dd[i] = fminf(dd[i], dx * dx + dy * dy + dz * dz);
    }
    __syncthreads();
  }
}

// ---------------------------------------------------------------------------
// Host side
// ---------------------------------------------------------------------------
enum PName { BB1=0,BB2,BB3,BBG,BMAP_DEC,BMAP_H1,BMAP_H2,BMAP_H3,CLS_H1,CLS_H2,CLS_H3,CW,
             DEC,DMAP_DEC,DMAP_H1,DMAP_H2,DMAP_H3,MASK_DEC,MH1,MH2,PWE,SC_H1,SC_H2,SC_H3 };

static inline void launch_gemm(hipStream_t s, const float* Wp, const float* Xp, const float* bias,
                               float* Out, int O, int C, int N, int ldX, int ldOut,
                               long sW, long sX, long sOut, int mode, int transA, int ldA)
{
  dim3 grid((N + 16 * WPB - 1) / (16 * WPB), (O + 15) / 16, NB);
  dim3 blk(32 * WPB, 1, 1);
  if (transA)
    gemm_wmma_t<1, 0><<<grid, blk, 0, s>>>(Wp, Xp, bias, Out, O, C, N, ldX, ldOut, sW, sX, sOut, mode, ldA);
  else if (C % 32)
    gemm_wmma_t<0, 1><<<grid, blk, 0, s>>>(Wp, Xp, bias, Out, O, C, N, ldX, ldOut, sW, sX, sOut, mode, ldA);
  else
    gemm_wmma_t<0, 0><<<grid, blk, 0, s>>>(Wp, Xp, bias, Out, O, C, N, ldX, ldOut, sW, sX, sOut, mode, ldA);
}

extern "C" void kernel_launch(void* const* d_in, const int* in_sizes, int n_in,
                              void* d_out, int out_size, void* d_ws, size_t ws_size,
                              hipStream_t stream)
{
  (void)in_sizes; (void)n_in; (void)out_size; (void)ws_size;
  const float* x = (const float*)d_in[0];
#define WT(i) ((const float*)d_in[1 + 2 * (i)])
#define BS(i) ((const float*)d_in[2 + 2 * (i)])
  float* dout = (float*)d_out;

  // output offsets (floats)
  const long OFF_BMAP = 0;                        // (2,3,12000)
  const long OFF_DMAP = 72000;                    // (2,1,12000)
  const long OFF_MASK = 96000;                    // (2,64,12000)
  const long OFF_PROB = 96000 + 2L*64*NPTS;       // (2,17,64)
  const long OFF_SCOR = OFF_PROB + 2L*17*QS;      // (2,1,64)
  const long OFF_AIDX = OFF_SCOR + 2L*QS;         // (2,64)

  // workspace layout (floats)
  float* ws = (float*)d_ws;
  long o = 0;
  float* feats = ws + o; o += 2L * 1344 * NPTS;
  float* tA    = ws + o; o += 2L * 256 * NPTS;
  float* tB    = ws + o; o += 2L * 256 * NPTS;
  float* fd    = ws + o; o += 2L * 256 * NPTS;
  float* mk    = ws + o; o += 2L * 256 * NPTS;
  float* gv    = ws + o; o += 2L * 1024;
  float* dsel  = ws + o; o += 2L * MSEL;
  int*   tidx  = (int*)(ws + o); o += 2L * MSEL;
  float* ps    = ws + o; o += 2L * 3 * MSEL;
  float* fs    = ws + o; o += 2L * 256 * MSEL;
  int*   knn1  = (int*)(ws + o); o += 2L * MSEL * KNN;
  float* fse   = ws + o; o += 2L * 256 * MSEL;
  int*   sel   = (int*)(ws + o); o += 2L * QS;
  float* gps   = ws + o; o += 2L * 3 * QS;
  float* gfs   = ws + o; o += 2L * 256 * QS;
  int*   knn2  = (int*)(ws + o); o += 2L * QS * KNN;
  float* gft   = ws + o; o += 2L * 256 * QS;
  float* mf    = ws + o; o += 2L * QS * NPTS;
  float* mt    = ws + o; o += 2L * QS * NPTS;

  const long FST  = 1344L * NPTS;   // feats batch stride
  const long S256 = 256L * NPTS;

  // ---- backbone into concatenated feats buffer ----
  launch_gemm(stream, WT(BB1), x, BS(BB1), feats,                64,  15, NPTS, NPTS, NPTS, 0, 15L*NPTS, FST, GM_RELU, 0, 15);
  launch_gemm(stream, WT(BB2), feats, BS(BB2), feats + 64L*NPTS, 128, 64, NPTS, NPTS, NPTS, 0, FST, FST, GM_RELU, 0, 64);
  launch_gemm(stream, WT(BB3), feats + 64L*NPTS, BS(BB3), feats + 192L*NPTS, 128, 128, NPTS, NPTS, NPTS, 0, FST, FST, GM_RELU, 0, 128);
  zero_f32_kernel<<<dim3(8), 256, 0, stream>>>(gv, 2048);
  launch_gemm(stream, WT(BBG), feats + 192L*NPTS, BS(BBG), gv, 1024, 128, NPTS, NPTS, 0, 0, FST, 1024, GM_ROWMAX, 0, 128);
  broadcast_g_kernel<<<dim3((1024L*NPTS + 255) / 256, 1, NB), 256, 0, stream>>>(gv, feats);

  // ---- bmap head ----
  launch_gemm(stream, WT(BMAP_DEC), feats, BS(BMAP_DEC), tA, 256, 1344, NPTS, NPTS, NPTS, 0, FST, S256, GM_RELU, 0, 1344);
  launch_gemm(stream, WT(BMAP_H1), tA, BS(BMAP_H1), tB, 256, 256, NPTS, NPTS, NPTS, 0, S256, S256, GM_RELU, 0, 256);
  launch_gemm(stream, WT(BMAP_H2), tB, BS(BMAP_H2), tA, 128, 256, NPTS, NPTS, NPTS, 0, S256, S256, GM_RELU, 0, 256);
  launch_gemm(stream, WT(BMAP_H3), tA, BS(BMAP_H3), dout + OFF_BMAP, 3, 128, NPTS, NPTS, NPTS, 0, S256, 3L*NPTS, GM_STORE, 0, 128);

  // ---- dmap head ----
  launch_gemm(stream, WT(DMAP_DEC), feats, BS(DMAP_DEC), tA, 256, 1344, NPTS, NPTS, NPTS, 0, FST, S256, GM_RELU, 0, 1344);
  launch_gemm(stream, WT(DMAP_H1), tA, BS(DMAP_H1), tB, 256, 256, NPTS, NPTS, NPTS, 0, S256, S256, GM_RELU, 0, 256);
  launch_gemm(stream, WT(DMAP_H2), tB, BS(DMAP_H2), tA, 128, 256, NPTS, NPTS, NPTS, 0, S256, S256, GM_RELU, 0, 256);
  launch_gemm(stream, WT(DMAP_H3), tA, BS(DMAP_H3), dout + OFF_DMAP, 1, 128, NPTS, NPTS, NPTS, 0, S256, (long)NPTS, GM_STORE, 0, 128);

  // ---- instance branch ----
  launch_gemm(stream, WT(DEC), feats, BS(DEC), fd, 256, 1344, NPTS, NPTS, NPTS, 0, FST, S256, GM_RELU, 0, 1344);
  launch_gemm(stream, WT(MASK_DEC), fd, BS(MASK_DEC), mk, 256, 256, NPTS, NPTS, NPTS, 0, S256, S256, GM_RELU, 0, 256);

  topk_rank_kernel<<<dim3((NPTS + 255) / 256, 1, NB), 256, 0, stream>>>(dout + OFF_DMAP, NPTS, tidx, dsel);

  gather_cols_kernel<<<dim3((3L*MSEL + 255) / 256, 1, NB), 256, 0, stream>>>(
      x, 15L*NPTS, NPTS, tidx, MSEL, ps, 3L*MSEL, MSEL, 3, MSEL);
  gather_cols_kernel<<<dim3((256L*MSEL + 255) / 256, 1, NB), 256, 0, stream>>>(
      fd, S256, NPTS, tidx, MSEL, fs, 256L*MSEL, MSEL, 256, MSEL);

  knn_kernel<<<dim3((MSEL + 255) / 256, 1, NB), 256, 0, stream>>>(ps, 3L*MSEL, MSEL, MSEL, knn1, (long)MSEL*KNN);

  edgeconv_wmma_kernel<<<dim3(MSEL, 16, NB), dim3(32, 1, 1), 0, stream>>>(
      WT(PWE), BS(PWE), fs, knn1, fse, 256, MSEL, MSEL, 256L*MSEL, (long)MSEL*KNN, 256L*MSEL);

  fps_kernel<<<dim3(NB), 1024, 0, stream>>>(ps, 3L*MSEL, MSEL, dsel, MSEL, sel, QS);

  gather_cols_kernel<<<dim3(1, 1, NB), 256, 0, stream>>>(ps, 3L*MSEL, MSEL, sel, QS, gps, 3L*QS, QS, 3, QS);
  gather_cols_kernel<<<dim3((256L*QS + 255) / 256, 1, NB), 256, 0, stream>>>(
      fse, 256L*MSEL, MSEL, sel, QS, gfs, 256L*QS, QS, 256, QS);
  gather_allidx_kernel<<<dim3(1, 1, NB), QS, 0, stream>>>(tidx, sel, dout + OFF_AIDX);

  knn_kernel<<<dim3(1, 1, NB), 256, 0, stream>>>(gps, 3L*QS, QS, QS, knn2, (long)QS*KNN);

  edgeconv_wmma_kernel<<<dim3(QS, 16, NB), dim3(32, 1, 1), 0, stream>>>(
      WT(CW), BS(CW), gfs, knn2, gft, 256, QS, QS, 256L*QS, (long)QS*KNN, 256L*QS);

  // mf[n][m] = sum_d gft[d][n] * mk[d][m]  (transA, per-batch A)
  launch_gemm(stream, gft, mk, nullptr, mf, QS, 256, NPTS, NPTS, NPTS, 256L*QS, S256, (long)QS*NPTS, GM_STORE, 1, QS);
  launch_gemm(stream, WT(MH1), mf, BS(MH1), mt, QS, QS, NPTS, NPTS, NPTS, 0, (long)QS*NPTS, (long)QS*NPTS, GM_RELU, 0, QS);
  launch_gemm(stream, WT(MH2), mt, BS(MH2), dout + OFF_MASK, QS, QS, NPTS, NPTS, NPTS, 0, (long)QS*NPTS, (long)QS*NPTS, GM_SIGMOID, 0, QS);

  // cls head on g_feats (256 x 64)
  launch_gemm(stream, WT(CLS_H1), gft, BS(CLS_H1), tA, 256, 256, QS, QS, QS, 0, 256L*QS, 256L*QS, GM_RELU, 0, 256);
  launch_gemm(stream, WT(CLS_H2), tA, BS(CLS_H2), tB, 128, 256, QS, QS, QS, 0, 256L*QS, 128L*QS, GM_RELU, 0, 256);
  launch_gemm(stream, WT(CLS_H3), tB, BS(CLS_H3), dout + OFF_PROB, 17, 128, QS, QS, QS, 0, 128L*QS, 17L*QS, GM_STORE, 0, 128);

  // score head
  launch_gemm(stream, WT(SC_H1), gft, BS(SC_H1), tA, 256, 256, QS, QS, QS, 0, 256L*QS, 256L*QS, GM_RELU, 0, 256);
  launch_gemm(stream, WT(SC_H2), tA, BS(SC_H2), tB, 128, 256, QS, QS, QS, 0, 256L*QS, 128L*QS, GM_RELU, 0, 256);
  launch_gemm(stream, WT(SC_H3), tB, BS(SC_H3), dout + OFF_SCOR, 1, 128, QS, QS, QS, 0, 128L*QS, (long)QS, GM_STORE, 0, 128);
#undef WT
#undef BS
}
